// Qwen3AttentionFused_13383118095062
// MI455X (gfx1250) — matrix operand order repaired and checked by reading, not compiled, not run on hardware
//
#include <hip/hip_runtime.h>
#include <hip/hip_bf16.h>
#include <math.h>

// ---------------- problem constants ----------------
#define B_    2
#define S_    2048
#define H_    2048
#define NH_   16
#define NKV_  8
#define HD_   128
#define QKV_O ((NH_ + 2 * NKV_) * HD_)   // 4096
#define MROWS (B_ * S_)                  // 4096
#define EPS_   1e-6f
#define THETA_ 10000.0f
#define SCALE_ 0.08838834764831845f      // 128^-0.5

// ---------------- WMMA types ----------------
typedef __bf16 bf16_t;
typedef bf16_t bf16x16 __attribute__((ext_vector_type(16)));
typedef float  f32x8   __attribute__((ext_vector_type(8)));
typedef unsigned int u32x4 __attribute__((ext_vector_type(4)));
typedef unsigned int u32x8 __attribute__((ext_vector_type(8)));

union FragU { bf16x16 v; uint4 q[2]; };
union AccU  { f32x8 v; float f[8]; };

__device__ __forceinline__ unsigned short f2bf(float f) {
    unsigned int u = __float_as_uint(f);
    unsigned int r = u + 0x7FFFu + ((u >> 16) & 1u);   // round-to-nearest-even
    return (unsigned short)(r >> 16);
}

__device__ __forceinline__ f32x8 wmma_bf16(const FragU& a, const FragU& b, f32x8 c) {
    return __builtin_amdgcn_wmma_f32_16x16x32_bf16(false, a.v, false, b.v,
                                                   (short)0, c, false, false);
}

// ---- Tensor Data Mover: load a [rows x 32 bf16] tile (row-major, row stride
// Kelems) from global into LDS at lds_off, padding each 64B row to 80B so the
// LDS image matches the TLD=40 fragment-loader stride.
// D# packing per CDNA5 ISA.
__device__ __forceinline__ void tdm_load_tile32(unsigned lds_off,
                                                const unsigned short* gptr,
                                                unsigned rows,
                                                unsigned tensor_rows,
                                                unsigned Kelems) {
    unsigned long long ga = (unsigned long long)(size_t)gptr;
    u32x4 g0;
    g0[0] = 1u;                                   // count=1, user descriptor
    g0[1] = lds_off;                              // lds_addr (bytes)
    g0[2] = (unsigned)ga;                         // global_addr[31:0]
    g0[3] = (unsigned)(ga >> 32) | 0x80000000u;   // global_addr[56:32] | type=2
    u32x8 g1;
    // data_size=2B (code 1) | pad_enable | pad_interval=16 DW | pad_amount=4 DW
    g1[0] = (1u << 16) | (1u << 20) | (3u << 22) | (3u << 25);
    g1[1] = (Kelems & 0xFFFFu) << 16;                               // tensor_dim0 lo
    g1[2] = ((Kelems >> 16) & 0xFFFFu) | ((tensor_rows & 0xFFFFu) << 16); // dim0 hi | dim1 lo
    g1[3] = ((tensor_rows >> 16) & 0xFFFFu) | (32u << 16);          // dim1 hi | tile_dim0=32
    g1[4] = rows;                                                   // tile_dim1
    g1[5] = Kelems;                                                 // tensor_dim0_stride lo
    g1[6] = 0u;
    g1[7] = 0u;
    asm volatile("tensor_load_to_lds %0, %1" :: "s"(g0), "s"(g1) : "memory");
}

// ---------------- kernel 1: fp32 -> bf16 convert ----------------
__global__ void k_f32_to_bf16(const float* __restrict__ in,
                              unsigned short* __restrict__ out, int n) {
    int idx = (blockIdx.x * blockDim.x + threadIdx.x) * 4;
    if (idx < n) {
        float4 v = *(const float4*)(in + idx);
        unsigned short h[4] = { f2bf(v.x), f2bf(v.y), f2bf(v.z), f2bf(v.w) };
        *(uint2*)(out + idx) = *(const uint2*)h;
    }
}

// ---------------- kernel 2: RMSNorm -> bf16 ----------------
__global__ __launch_bounds__(256)
void k_rmsnorm_bf16(const float* __restrict__ x, const float* __restrict__ w,
                    unsigned short* __restrict__ out) {
    __shared__ float red[256];
    const int row = blockIdx.x;
    const int t = threadIdx.x;
    const float* xr = x + (size_t)row * H_;
    float4 a = ((const float4*)xr)[t * 2];
    float4 b = ((const float4*)xr)[t * 2 + 1];
    float ss = a.x*a.x + a.y*a.y + a.z*a.z + a.w*a.w +
               b.x*b.x + b.y*b.y + b.z*b.z + b.w*b.w;
    red[t] = ss;
    __syncthreads();
    for (int off = 128; off > 0; off >>= 1) {
        if (t < off) red[t] += red[t + off];
        __syncthreads();
    }
    float scale = rsqrtf(red[0] / (float)H_ + EPS_);
    const float* wr = w + t * 8;
    float v[8] = { a.x, a.y, a.z, a.w, b.x, b.y, b.z, b.w };
    unsigned short h[8];
#pragma unroll
    for (int i = 0; i < 8; i++) h[i] = f2bf(v[i] * scale * wr[i]);
    ((uint4*)(out + (size_t)row * H_))[t] = *(const uint4*)h;
}

// ---------------- kernel 3/6: bf16 WMMA GEMM  C[M,N] = A[M,K] * Bw[N,K]^T ----
// Block: 256 thr = 8 waves, tile 64(M) x 128(N), K-step 32.
// A/B tiles staged global->LDS by the Tensor Data Mover, DOUBLE-BUFFERED:
// waves 0/1 issue the TDM for tile i+1 before computing tile i, and drain
// TENSORcnt only after the WMMA work, so DMA overlaps matrix math.
#define TLD 40   // padded LDS row stride in bf16 elems (80B) == TDM pad config

template <bool ADD_RES>
__global__ __launch_bounds__(256)
void k_gemm_bf16(const unsigned short* __restrict__ A,
                 const unsigned short* __restrict__ Bw,
                 const float* __restrict__ Res,
                 float* __restrict__ C, int M, int N, int K) {
    __shared__ __align__(16) unsigned short sA[2][64 * TLD];
    __shared__ __align__(16) unsigned short sB[2][128 * TLD];
    const int t = threadIdx.x;
    const int lane = t & 31, wave = t >> 5;
    const int lm = lane & 15, lh = lane >> 4;
    const int wm = (wave >> 2) * 32;
    const int wn = (wave & 3) * 32;
    const int bm = blockIdx.y * 64;
    const int bn = blockIdx.x * 128;
    const unsigned ldsA[2] = { (unsigned)(size_t)&sA[0][0], (unsigned)(size_t)&sA[1][0] };
    const unsigned ldsB[2] = { (unsigned)(size_t)&sB[0][0], (unsigned)(size_t)&sB[1][0] };
    const unsigned short* Abase = A + (size_t)bm * K;
    const unsigned short* Bbase = Bw + (size_t)bn * K;

    const f32x8 zero8 = { 0.f,0.f,0.f,0.f,0.f,0.f,0.f,0.f };
    f32x8 acc[2][2];
    acc[0][0] = zero8; acc[0][1] = zero8; acc[1][0] = zero8; acc[1][1] = zero8;

    // prologue: stage tile 0 into buffer 0
    if (wave == 0) {
        tdm_load_tile32(ldsA[0], Abase, 64u, (unsigned)M, (unsigned)K);
        __builtin_amdgcn_s_wait_tensorcnt(0);
    } else if (wave == 1) {
        tdm_load_tile32(ldsB[0], Bbase, 128u, (unsigned)N, (unsigned)K);
        __builtin_amdgcn_s_wait_tensorcnt(0);
    }
    __syncthreads();

    for (int k0 = 0; k0 < K; k0 += 32) {
        const int cur = (k0 >> 5) & 1;
        const bool more = (k0 + 32) < K;
        if (more) {   // async prefetch of next tile into the other buffer
            if (wave == 0)
                tdm_load_tile32(ldsA[cur ^ 1], Abase + k0 + 32, 64u,
                                (unsigned)M, (unsigned)K);
            else if (wave == 1)
                tdm_load_tile32(ldsB[cur ^ 1], Bbase + k0 + 32, 128u,
                                (unsigned)N, (unsigned)K);
        }

        FragU a0, a1, b0, b1;
        {   // A fragment: lane-half holds K = {8*lh..} and {16+8*lh..}
            const unsigned short* p = sA[cur] + (wm + lm) * TLD;
            a0.q[0] = *(const uint4*)(p + 8 * lh);
            a0.q[1] = *(const uint4*)(p + 16 + 8 * lh);
            p += 16 * TLD;
            a1.q[0] = *(const uint4*)(p + 8 * lh);
            a1.q[1] = *(const uint4*)(p + 16 + 8 * lh);
        }
        {   // B fragment: lane-half holds K = {8*lh..} and {16+8*lh..}
            const unsigned short* p = sB[cur] + (wn + lm) * TLD;
            b0.q[0] = *(const uint4*)(p + 8 * lh);
            b0.q[1] = *(const uint4*)(p + 16 + 8 * lh);
            p += 16 * TLD;
            b1.q[0] = *(const uint4*)(p + 8 * lh);
            b1.q[1] = *(const uint4*)(p + 16 + 8 * lh);
        }
        acc[0][0] = wmma_bf16(a0, b0, acc[0][0]);
        acc[0][1] = wmma_bf16(a0, b1, acc[0][1]);
        acc[1][0] = wmma_bf16(a1, b0, acc[1][0]);
        acc[1][1] = wmma_bf16(a1, b1, acc[1][1]);

        if (more && wave <= 1) __builtin_amdgcn_s_wait_tensorcnt(0);
        __syncthreads();   // next buffer ready + current buffer fully consumed
    }

#pragma unroll
    for (int mi = 0; mi < 2; mi++)
#pragma unroll
        for (int ni = 0; ni < 2; ni++) {
            AccU u; u.v = acc[mi][ni];
            int col = bn + wn + ni * 16 + lm;
#pragma unroll
            for (int r = 0; r < 8; r++) {
                int row = bm + wm + mi * 16 + r + 8 * lh;
                size_t idx = (size_t)row * N + col;
                C[idx] = ADD_RES ? (u.f[r] + Res[idx]) : u.f[r];
            }
        }
}

// ---------------- kernel 4: per-head q/k RMSNorm + RoPE, split to bf16 -------
__global__ __launch_bounds__(128)
void k_qk_post(const float* __restrict__ qkv,       // [MROWS][QKV_O]
               const int* __restrict__ positions,   // [B*S]
               const float* __restrict__ qw, const float* __restrict__ kw,
               unsigned short* __restrict__ qout,   // [B,NH,S,HD]
               unsigned short* __restrict__ kout,   // [B,NKV,S,HD]
               unsigned short* __restrict__ vout) { // [B,NKV,S,HD]
    __shared__ float sred[HD_];
    __shared__ float sbuf[HD_];
    const int d = threadIdx.x;
    const int row = blockIdx.x;          // b*S + s
    const int hid = blockIdx.y;          // 0..31
    const int b = row / S_, s = row - b * S_;
    float val = qkv[(size_t)row * QKV_O + hid * HD_ + d];
    if (hid < NH_ + NKV_) {
        sred[d] = val * val;
        __syncthreads();
        for (int off = 64; off > 0; off >>= 1) {
            if (d < off) sred[d] += sred[d + off];
            __syncthreads();
        }
        float scale = rsqrtf(sred[0] / (float)HD_ + EPS_);
        const float* w = (hid < NH_) ? qw : kw;
        sbuf[d] = val * scale * w[d];
        __syncthreads();
        int pos = positions[row];
        int fi = d & 63;
        float inv = __expf(-(2.0f * (float)fi / (float)HD_) * __logf(THETA_));
        float sn, cs;
        __sincosf((float)pos * inv, &sn, &cs);
        float outv = (d < 64) ? (sbuf[d] * cs - sbuf[d + 64] * sn)
                              : (sbuf[d] * cs + sbuf[d - 64] * sn);
        if (hid < NH_)
            qout[(((size_t)b * NH_ + hid) * S_ + s) * HD_ + d] = f2bf(outv);
        else
            kout[(((size_t)b * NKV_ + (hid - NH_)) * S_ + s) * HD_ + d] = f2bf(outv);
    } else {
        vout[(((size_t)b * NKV_ + (hid - NH_ - NKV_)) * S_ + s) * HD_ + d] = f2bf(val);
    }
}

// ---------------- kernel 5: V transpose [B,NKV,S,HD] -> [B,NKV,HD,S] ---------
__global__ __launch_bounds__(256)
void k_transpose_v(const unsigned short* __restrict__ vin,
                   unsigned short* __restrict__ vt) {
    __shared__ unsigned short tile[32][33];
    const int bh = blockIdx.z;
    const unsigned short* src = vin + (size_t)bh * S_ * HD_;
    unsigned short* dst = vt + (size_t)bh * S_ * HD_;
    const int s0 = blockIdx.x * 32, d0 = blockIdx.y * 32;
    const int tx = threadIdx.x, ty = threadIdx.y;  // 32 x 8
#pragma unroll
    for (int i = 0; i < 32; i += 8)
        tile[ty + i][tx] = src[(size_t)(s0 + ty + i) * HD_ + d0 + tx];
    __syncthreads();
#pragma unroll
    for (int i = 0; i < 32; i += 8)
        dst[(size_t)(d0 + ty + i) * S_ + s0 + tx] = tile[tx][ty + i];
}

// ---------------- kernel 6: causal GQA flash attention (WMMA) ----------------
// Block = 4 waves; wave owns 16 q rows; KV tiles of 32.
// K and V^T B-fragments are loaded straight from global (K-contiguous rows).
__global__ __launch_bounds__(128)
void k_flash_attn(const unsigned short* __restrict__ Q,   // [B,NH,S,HD]
                  const unsigned short* __restrict__ Kc,  // [B,NKV,S,HD]
                  const unsigned short* __restrict__ VT,  // [B,NKV,HD,S]
                  unsigned short* __restrict__ O) {       // [B*S, NH*HD]
    __shared__ __align__(16) unsigned short sP[4][16 * TLD];   // per-wave P
    const int t = threadIdx.x;
    const int lane = t & 31, wave = t >> 5;
    const int lm = lane & 15, lh = lane >> 4;
    const int h = blockIdx.y, b = blockIdx.z;
    const int q0 = blockIdx.x * 64 + wave * 16;
    const size_t qbase  = ((size_t)b * NH_ + h) * S_ * HD_;
    const size_t kvbase = ((size_t)b * NKV_ + (h >> 1)) * S_ * HD_;

    FragU qa[4];
    {
        const unsigned short* p = Q + qbase + (size_t)(q0 + lm) * HD_;
#pragma unroll
        for (int j = 0; j < 4; j++) {
            qa[j].q[0] = *(const uint4*)(p + j * 32 + 8 * lh);
            qa[j].q[1] = *(const uint4*)(p + j * 32 + 16 + 8 * lh);
        }
    }

    const f32x8 zero8 = { 0.f,0.f,0.f,0.f,0.f,0.f,0.f,0.f };
    float mrow[8], lrow[8];
    f32x8 accO[8];
#pragma unroll
    for (int r = 0; r < 8; r++) { mrow[r] = -1e30f; lrow[r] = 0.f; }
#pragma unroll
    for (int n = 0; n < 8; n++) accO[n] = zero8;

    unsigned short* Pb = sP[wave];
    const int tmax = (q0 + 15) >> 5;

    for (int tix = 0; tix <= tmax; ++tix) {
        const int k0 = tix * 32;
        // ---- scores S = Q * K^T (16 x 32) ----
        f32x8 sacc[2]; sacc[0] = zero8; sacc[1] = zero8;
#pragma unroll
        for (int nt = 0; nt < 2; ++nt) {
            const unsigned short* kp =
                Kc + kvbase + (size_t)(k0 + nt * 16 + lm) * HD_ + 8 * lh;
#pragma unroll
            for (int j = 0; j < 4; j++) {
                FragU kb;
                kb.q[0] = *(const uint4*)(kp + j * 32);
                kb.q[1] = *(const uint4*)(kp + j * 32 + 16);
                sacc[nt] = wmma_bf16(qa[j], kb, sacc[nt]);
            }
        }
        // ---- online softmax (C layout: N=lm, M=r+8*lh) ----
        AccU s0, s1; s0.v = sacc[0]; s1.v = sacc[1];
        float alpha[8];
#pragma unroll
        for (int r = 0; r < 8; r++) {
            int qrow = q0 + r + 8 * lh;
            float v0 = (k0 + lm      <= qrow) ? s0.f[r] * SCALE_ : -__builtin_inff();
            float v1 = (k0 + 16 + lm <= qrow) ? s1.f[r] * SCALE_ : -__builtin_inff();
            float mloc = fmaxf(v0, v1);
            for (int msk = 1; msk < 16; msk <<= 1)
                mloc = fmaxf(mloc, __shfl_xor(mloc, msk, 32));
            float mnew = fmaxf(mrow[r], mloc);
            alpha[r] = __expf(mrow[r] - mnew);
            mrow[r] = mnew;
            float p0 = __expf(v0 - mnew);
            float p1 = __expf(v1 - mnew);
            s0.f[r] = p0; s1.f[r] = p1;
            float ls = p0 + p1;
            for (int msk = 1; msk < 16; msk <<= 1)
                ls += __shfl_xor(ls, msk, 32);
            lrow[r] = lrow[r] * alpha[r] + ls;
        }
#pragma unroll
        for (int n = 0; n < 8; n++) {
            AccU u; u.v = accO[n];
#pragma unroll
            for (int r = 0; r < 8; r++) u.f[r] *= alpha[r];
            accO[n] = u.v;
        }
        // ---- P: C-layout -> LDS -> A-layout fragment ----
#pragma unroll
        for (int r = 0; r < 8; r++) {
            int m = r + 8 * lh;
            Pb[m * TLD + lm]      = f2bf(s0.f[r]);
            Pb[m * TLD + 16 + lm] = f2bf(s1.f[r]);
        }
        asm volatile("s_wait_dscnt 0" ::: "memory");  // cross-lane LDS RAW
        FragU pa;
        {
            const unsigned short* p = Pb + lm * TLD;
            pa.q[0] = *(const uint4*)(p + 8 * lh);
            pa.q[1] = *(const uint4*)(p + 16 + 8 * lh);
        }
        // ---- O += P * V : V^T rows are kv-contiguous in global ----
#pragma unroll
        for (int n = 0; n < 8; n++) {
            FragU vb;
            const unsigned short* p =
                VT + kvbase + (size_t)(n * 16 + lm) * S_ + k0 + 8 * lh;
            vb.q[0] = *(const uint4*)(p);
            vb.q[1] = *(const uint4*)(p + 16);
            accO[n] = wmma_bf16(pa, vb, accO[n]);
        }
    }

    // ---- normalize and store bf16 attn output [B*S, NH*HD] ----
    float invl[8];
#pragma unroll
    for (int r = 0; r < 8; r++) invl[r] = 1.0f / lrow[r];
#pragma unroll
    for (int n = 0; n < 8; n++) {
        AccU u; u.v = accO[n];
#pragma unroll
        for (int r = 0; r < 8; r++) {
            int qrow = q0 + r + 8 * lh;
            size_t idx = ((size_t)b * S_ + qrow) * (size_t)(NH_ * HD_)
                         + h * HD_ + n * 16 + lm;
            O[idx] = f2bf(u.f[r] * invl[r]);
        }
    }
}

// ---------------- host-side orchestration ----------------
extern "C" void kernel_launch(void* const* d_in, const int* in_sizes, int n_in,
                              void* d_out, int out_size, void* d_ws, size_t ws_size,
                              hipStream_t stream) {
    const int*   positions = (const int*)d_in[0];
    const float* hidden    = (const float*)d_in[1];
    const float* residual  = (const float*)d_in[2];
    const float* ln_w      = (const float*)d_in[3];
    const float* qkv_w     = (const float*)d_in[4];
    const float* o_w       = (const float*)d_in[5];
    const float* q_norm_w  = (const float*)d_in[6];
    const float* k_norm_w  = (const float*)d_in[7];
    float* out = (float*)d_out;

    // workspace layout (aliased across pipeline phases), ~104 MB total:
    //  A [16MB] @  0: xn bf16        -> later q bf16
    //  B [16MB] @ 16: wqkv bf16      -> later k bf16 (8MB) + v bf16 (8MB)
    //  C [ 8MB] @ 32: wo bf16        (persists)
    //  D [64MB] @ 40: qkv f32        -> later attn bf16 (16MB) + v^T bf16 (8MB @ +16MB)
    char* ws = (char*)d_ws;
    unsigned short* xn   = (unsigned short*)(ws);
    unsigned short* wqkv = (unsigned short*)(ws + (size_t)(16u << 20));
    unsigned short* wo   = (unsigned short*)(ws + (size_t)(32u << 20));
    float*          qkv  = (float*)(ws + (size_t)(40u << 20));
    unsigned short* qbf  = xn;
    unsigned short* kbf  = wqkv;
    unsigned short* vbf  = wqkv + (size_t)B_ * NKV_ * S_ * HD_;
    unsigned short* attn = (unsigned short*)qkv;
    unsigned short* vtb  = (unsigned short*)(ws + (size_t)(56u << 20));

    k_f32_to_bf16<<<(QKV_O * H_ / 4) / 256, 256, 0, stream>>>(qkv_w, wqkv, QKV_O * H_);
    k_f32_to_bf16<<<(H_ * H_ / 4) / 256, 256, 0, stream>>>(o_w, wo, H_ * H_);
    k_rmsnorm_bf16<<<MROWS, 256, 0, stream>>>(hidden, ln_w, xn);
    k_gemm_bf16<false><<<dim3(QKV_O / 128, MROWS / 64), 256, 0, stream>>>(
        xn, wqkv, nullptr, qkv, MROWS, QKV_O, H_);
    k_qk_post<<<dim3(MROWS, NH_ + 2 * NKV_), HD_, 0, stream>>>(
        qkv, positions, q_norm_w, k_norm_w, qbf, kbf, vbf);
    k_transpose_v<<<dim3(S_ / 32, HD_ / 32, B_ * NKV_), dim3(32, 8), 0, stream>>>(
        vbf, vtb);
    k_flash_attn<<<dim3(S_ / 64, NH_, B_), 128, 0, stream>>>(qbf, kbf, vtb, attn);
    k_gemm_bf16<true><<<dim3(H_ / 128, MROWS / 64), 256, 0, stream>>>(
        attn, wo, residual, out, MROWS, H_, H_);
}
